// EnhancedTimeSeriesModel_13125420057130
// MI455X (gfx1250) — matrix operand order repaired and checked by reading, not compiled, not run on hardware
//
#include <hip/hip_runtime.h>
#include <hip/hip_bf16.h>

// ---------------------------------------------------------------------------
// EnhancedTimeSeriesModel forward for MI455X (gfx1250, wave32, WMMA)
// All dense contractions -> v_wmma_f32_16x16x32_bf16 (fp32 accumulate).
// Global->LDS B-operand traffic uses async-to-LDS DMA (ASYNCcnt) with a
// double-buffered LDS pipeline so WMMA overlaps the streaming loads.
// ---------------------------------------------------------------------------

#define B_    64
#define L_    256
#define D_    512
#define H_    8
#define P_    24
#define F_    16
#define NL_   4
#define DK_   64
#define DFF_  2048
#define S_    1024          // 4*L sampled keys
#define LP_   (L_ + 6)      // halo 3 each side (covers k3/k5/k7)
#define ROWS_ (B_ * L_)

typedef __attribute__((ext_vector_type(16))) __bf16 v16bf;
typedef __attribute__((ext_vector_type(8)))  float  v8f;
typedef __bf16 bf16_t;

#define GF_ACCUM 1
#define GF_BIAS  2
#define GF_GELU  4
#define GF_BF16  8          // also emit bf16 copy of C

__device__ __forceinline__ float gelu_exact(float x) {
  return 0.5f * x * (1.0f + erff(x * 0.7071067811865475f));
}

// ---- CDNA5 async-to-LDS helpers ------------------------------------------
// 16B per lane, LDS dest = low 32 bits of generic shared pointer (flat rule:
// LDS aperture addr[31:0] is the wave-relative LDS byte offset).
__device__ __forceinline__ void async_ld16(unsigned lds_off, const void* gaddr) {
  asm volatile("global_load_async_to_lds_b128 %0, %1, off"
               :: "v"(lds_off), "v"(gaddr) : "memory");
}
__device__ __forceinline__ void wait_async2() {
#if __has_builtin(__builtin_amdgcn_s_wait_asynccnt)
  __builtin_amdgcn_s_wait_asynccnt(2);
#else
  asm volatile("s_wait_asynccnt 2" ::: "memory");
#endif
}
__device__ __forceinline__ void wait_async0() {
#if __has_builtin(__builtin_amdgcn_s_wait_asynccnt)
  __builtin_amdgcn_s_wait_asynccnt(0);
#else
  asm volatile("s_wait_asynccnt 0" ::: "memory");
#endif
}

// ---------------------------------------------------------------------------
// Generic WMMA GEMM: C[M,N] (=|+=) rowmap(A)[M,K](fp32) * B[K,N](bf16) (+bias)(gelu)
// Block: 256 thr = 8 waves; tile 128x128; wave = 64Mx32N = 4x2 WMMA accums.
// B slabs stream in via async-to-LDS, double-buffered (s_wait_asynccnt 2).
// Row map (conv taps / slicing): ar = (r/aOuter)*aLp + r%aOuter + aOff.
// ---------------------------------------------------------------------------
template<int FLAGS>
__global__ __launch_bounds__(256)
void wmma_gemm_kernel(const float* __restrict__ A, const bf16_t* __restrict__ Bm,
                      float* __restrict__ C, const float* __restrict__ bias,
                      bf16_t* __restrict__ Cbf,
                      int M, int N, int K, int lda, int ldb, int ldc,
                      int aOuter, int aLp, int aOff)
{
  __shared__ __align__(16) bf16_t As[2][128][40];    // 128x32 (+8 pad), x2
  __shared__ __align__(16) bf16_t Bs[2][32][136];    // 32x128 (+8 pad), x2 (272B rows, 16B aligned)

  const int tid  = threadIdx.x;
  const int lane = tid & 31;
  const int wid  = tid >> 5;
  const int lo   = lane & 15;
  const int hi   = lane >> 4;
  const int wm   = (wid & 1) * 64;
  const int wn   = (wid >> 1) * 32;
  const int tileM = blockIdx.y * 128;
  const int tileN = blockIdx.x * 128;
  const int nk = (K + 31) >> 5;

  // async B-tile issue: 2 x b128 per thread = 8KB tile; edge cells fall back
  // to predicated scalar fills (async must not touch OOB global memory).
  auto issueB = [&](int buf, int k0) {
#pragma unroll
    for (int e = 0; e < 2; ++e) {
      int li  = (tid << 1) + e;     // 0..511
      int row = li >> 4;            // 32 rows
      int col = (li & 15) << 3;     // 16B segments (8 bf16)
      if ((k0 + row) < K && (tileN + col + 7) < N) {
        async_ld16((unsigned)(size_t)&Bs[buf][row][col],
                   Bm + (size_t)(k0 + row) * ldb + tileN + col);
      } else {
#pragma unroll
        for (int c = 0; c < 8; ++c) {
          bf16_t v = (bf16_t)0.0f;
          if ((k0 + row) < K && (tileN + col + c) < N)
            v = Bm[(size_t)(k0 + row) * ldb + tileN + col + c];
          Bs[buf][row][col + c] = v;
        }
      }
    }
  };

  v8f acc[4][2];
#pragma unroll
  for (int i = 0; i < 4; ++i)
#pragma unroll
    for (int j = 0; j < 2; ++j)
      acc[i][j] = (v8f){0.f,0.f,0.f,0.f,0.f,0.f,0.f,0.f};

  issueB(0, 0);
  for (int kt = 0; kt < nk; ++kt) {
    const int k0  = kt << 5;
    const int cur = kt & 1;

    // ---- stage A (fp32 -> bf16), zero-pad OOB ----
#pragma unroll 4
    for (int idx = tid; idx < 128 * 32; idx += 256) {
      int r = idx >> 5, c = idx & 31;
      int gr = tileM + r;
      float v = 0.0f;
      if (gr < M && (k0 + c) < K) {
        int ar = aLp ? (gr / aOuter) * aLp + (gr % aOuter) + aOff : gr;
        v = A[(long)ar * lda + (k0 + c)];
      }
      As[cur][r][c] = (bf16_t)v;
    }

    if (kt + 1 < nk) {
      issueB(cur ^ 1, k0 + 32);     // stream next slab while we compute
      // prefetch next A slab into L2 (global_prefetch_b8, speculative)
      {
        int r = tid >> 1, c = (tid & 1) << 4;
        int gr = tileM + r;
        if (gr < M && (k0 + 32 + c) < K) {
          int ar = aLp ? (gr / aOuter) * aLp + (gr % aOuter) + aOff : gr;
          __builtin_prefetch(&A[(long)ar * lda + k0 + 32 + c], 0, 1);
        }
      }
      wait_async2();                // drain current slab, keep next in flight
    } else {
      wait_async0();
    }
    __syncthreads();

    // ---- B fragments: B[k][n], lane holds N=lo, K = 8*hi + e (+8 for e>=8) ----
    v16bf bfrag[2];
#pragma unroll
    for (int j = 0; j < 2; ++j) {
      int n0 = wn + j * 16;
#pragma unroll
      for (int e = 0; e < 16; ++e)
        bfrag[j][e] = Bs[cur][(e < 8) ? (hi * 8 + e) : (hi * 8 + e + 8)][n0 + lo];
    }
#pragma unroll
    for (int i = 0; i < 4; ++i) {
      int m0 = wm + i * 16;
      // A fragment: lane holds M=lo, K = 8*hi + e (+8 for e>=8)
      v16bf afrag;
#pragma unroll
      for (int e = 0; e < 16; ++e) {
        int kk = (e < 8) ? (hi * 8 + e) : (hi * 8 + e + 8);
        afrag[e] = As[cur][m0 + lo][kk];
      }
#pragma unroll
      for (int j = 0; j < 2; ++j)
        acc[i][j] = __builtin_amdgcn_wmma_f32_16x16x32_bf16(
            false, afrag, false, bfrag[j], (short)0, acc[i][j], false, false);
    }
    __syncthreads();                // protect buffers before next overwrite
  }

  // ---- epilogue: C layout M = 8*hi + r, N = lo ----
#pragma unroll
  for (int i = 0; i < 4; ++i)
#pragma unroll
    for (int j = 0; j < 2; ++j)
#pragma unroll
      for (int r = 0; r < 8; ++r) {
        int row = tileM + wm + i * 16 + hi * 8 + r;
        int col = tileN + wn + j * 16 + lo;
        if (row < M && col < N) {
          float v = acc[i][j][r];
          if (FLAGS & GF_ACCUM) v += C[(long)row * ldc + col];
          if (FLAGS & GF_BIAS)  v += bias[col];
          if (FLAGS & GF_GELU)  v = gelu_exact(v);
          C[(long)row * ldc + col] = v;
          if (FLAGS & GF_BF16)  Cbf[(long)row * ldc + col] = (bf16_t)v;
        }
      }
}

// ---------------------------------------------------------------------------
// Probsparse scoring: for each (b,h), M[l] = max_s(qk) - mean_s(qk) over the
// S=1024 sampled keys; WMMA 16x16 score tiles, qk never materialized.
// Q tile and sampled-K chunks are async-gathered from bf16 copies
// (per-lane addresses => the sample_idx gather maps directly onto async DMA).
// ---------------------------------------------------------------------------
__global__ __launch_bounds__(256)
void score_kernel(const bf16_t* __restrict__ Qbf, const bf16_t* __restrict__ Kbf,
                  const int* __restrict__ sidx, float* __restrict__ Mval)
{
  __shared__ __align__(16) bf16_t Qs[L_][DK_ + 8];      // 256x64 (+8 pad), 144B rows
  __shared__ __align__(16) bf16_t Ks[2][64][DK_ + 8];   // 64-key chunk, double buffered

  const int bh = blockIdx.x;
  const int b = bh / H_, h = bh % H_;
  const int tid = threadIdx.x, lane = tid & 31, wid = tid >> 5;
  const int lo = lane & 15, hi = lane >> 4;
  const float scale = 0.125f;          // 1/sqrt(64)
  const int NC = S_ / 64;

  // async-stage whole Q tile: one row per thread, 8 x 16B segments
  {
    const bf16_t* qp = Qbf + ((size_t)b * L_ + tid) * D_ + h * DK_;
#pragma unroll
    for (int e = 0; e < 8; ++e)
      async_ld16((unsigned)(size_t)&Qs[tid][e * 8], qp + e * 8);
  }
  auto issueK = [&](int buf, int sc) {
#pragma unroll
    for (int e = 0; e < 2; ++e) {
      int li  = (tid << 1) + e;       // 0..511
      int row = li >> 3, seg = li & 7;
      int ls = sidx[sc * 64 + row];
      async_ld16((unsigned)(size_t)&Ks[buf][row][seg * 8],
                 Kbf + ((size_t)b * L_ + ls) * D_ + h * DK_ + seg * 8);
    }
  };

  float rmax[2][8], rsum[2][8];
#pragma unroll
  for (int t = 0; t < 2; ++t)
#pragma unroll
    for (int r = 0; r < 8; ++r) { rmax[t][r] = -3.0e38f; rsum[t][r] = 0.0f; }

  issueK(0, 0);
  for (int sc = 0; sc < NC; ++sc) {
    const int cur = sc & 1;
    if (sc + 1 < NC) { issueK(cur ^ 1, sc + 1); wait_async2(); }
    else             { wait_async0(); }
    __syncthreads();

#pragma unroll
    for (int t = 0; t < 2; ++t) {        // 2 l-tiles per wave (8 waves * 2 * 16 = 256)
      int lt = (wid * 2 + t) * 16;
#pragma unroll
      for (int st = 0; st < 4; ++st) {
        v8f acc = (v8f){0.f,0.f,0.f,0.f,0.f,0.f,0.f,0.f};
#pragma unroll
        for (int kh = 0; kh < 2; ++kh) { // K = 64 = 2 x 32
          v16bf af, bf;
#pragma unroll
          for (int e = 0; e < 16; ++e) {
            int kk = ((e < 8) ? (hi * 8 + e) : (hi * 8 + e + 8)) + kh * 32;
            af[e] = Qs[lt + lo][kk];
            bf[e] = Ks[cur][st * 16 + lo][kk];
          }
          acc = __builtin_amdgcn_wmma_f32_16x16x32_bf16(
              false, af, false, bf, (short)0, acc, false, false);
        }
#pragma unroll
        for (int r = 0; r < 8; ++r) {
          float v = acc[r] * scale;
          rmax[t][r] = fmaxf(rmax[t][r], v);
          rsum[t][r] += v;
        }
      }
    }
    __syncthreads();
  }
  // reduce across the 16 lanes of each half-wave (each lane owns s%16==lo cols)
#pragma unroll
  for (int t = 0; t < 2; ++t)
#pragma unroll
    for (int r = 0; r < 8; ++r) {
      float m = rmax[t][r], s = rsum[t][r];
      for (int off = 1; off < 16; off <<= 1) {
        m = fmaxf(m, __shfl_xor(m, off, 32));
        s += __shfl_xor(s, off, 32);
      }
      if (lo == 0) {
        int l = (wid * 2 + t) * 16 + hi * 8 + r;
        Mval[(long)bh * L_ + l] = m - s * (1.0f / S_);
      }
    }
}

// ---------------------------------------------------------------------------
// argmax(M) -> qr; softmax(qr . K^T) . V -> ctx[b, h*DK+dk]  (tiny, VALU)
// ---------------------------------------------------------------------------
__global__ __launch_bounds__(256)
void attend_kernel(const float* __restrict__ Mval, const float* __restrict__ Q,
                   const float* __restrict__ Kmat, const float* __restrict__ V,
                   float* __restrict__ ctxB)
{
  __shared__ float sv[256];
  __shared__ int   si[256];
  __shared__ float qr[DK_];
  __shared__ float attn[L_];
  __shared__ float red[256];

  const int bh = blockIdx.x;
  const int b = bh / H_, h = bh % H_;
  const int tid = threadIdx.x;
  const float scale = 0.125f;

  // argmax over l (first-max tie-break, matches jnp.argmax)
  sv[tid] = Mval[(long)bh * L_ + tid]; si[tid] = tid;
  __syncthreads();
  for (int s = 128; s > 0; s >>= 1) {
    if (tid < s) {
      float a = sv[tid], c = sv[tid + s];
      if (c > a || (c == a && si[tid + s] < si[tid])) { sv[tid] = c; si[tid] = si[tid + s]; }
    }
    __syncthreads();
  }
  int U = si[0];
  if (tid < DK_) qr[tid] = Q[((long)b * L_ + U) * D_ + h * DK_ + tid];
  __syncthreads();

  float sc = 0.0f;
  const float* kp = Kmat + ((long)b * L_ + tid) * D_ + h * DK_;
#pragma unroll 8
  for (int d = 0; d < DK_; ++d) sc += qr[d] * kp[d];
  sc *= scale;

  red[tid] = sc; __syncthreads();
  for (int s = 128; s > 0; s >>= 1) { if (tid < s) red[tid] = fmaxf(red[tid], red[tid + s]); __syncthreads(); }
  float mx = red[0]; __syncthreads();
  float e = expf(sc - mx);
  red[tid] = e; __syncthreads();
  for (int s = 128; s > 0; s >>= 1) { if (tid < s) red[tid] += red[tid + s]; __syncthreads(); }
  float denom = red[0]; __syncthreads();
  attn[tid] = e / denom;
  __syncthreads();

  if (tid < DK_) {
    float c = 0.0f;
    const float* vp = V + (long)b * L_ * D_ + h * DK_ + tid;
    for (int l = 0; l < L_; ++l) c += attn[l] * vp[(long)l * D_];
    ctxB[(long)b * D_ + h * DK_ + tid] = c;
  }
}

// ---------------------------------------------------------------------------
// LayerNorm(h + res) in place; mode 1: res row-indexed, mode 2: res[b] bcast
// ---------------------------------------------------------------------------
__global__ __launch_bounds__(256)
void ln_kernel(float* __restrict__ h, const float* __restrict__ res,
               const float* __restrict__ w, const float* __restrict__ bia, int mode)
{
  __shared__ float red[256];
  const long row = blockIdx.x;
  const int tid = threadIdx.x;
  const float* rp = res + ((mode == 2) ? (row / L_) * (long)D_ : row * (long)D_);
  float x0 = h[row * D_ + tid]       + rp[tid];
  float x1 = h[row * D_ + tid + 256] + rp[tid + 256];
  red[tid] = x0 + x1; __syncthreads();
  for (int s = 128; s > 0; s >>= 1) { if (tid < s) red[tid] += red[tid + s]; __syncthreads(); }
  float mean = red[0] * (1.0f / D_); __syncthreads();
  float d0 = x0 - mean, d1 = x1 - mean;
  red[tid] = d0 * d0 + d1 * d1; __syncthreads();
  for (int s = 128; s > 0; s >>= 1) { if (tid < s) red[tid] += red[tid + s]; __syncthreads(); }
  float rstd = rsqrtf(red[0] * (1.0f / D_) + 1e-5f);
  h[row * D_ + tid]       = d0 * rstd * w[tid]       + bia[tid];
  h[row * D_ + tid + 256] = d1 * rstd * w[tid + 256] + bia[tid + 256];
}

// ---------------------------------------------------------------------------
// misc elementwise kernels
// ---------------------------------------------------------------------------
__global__ __launch_bounds__(256)
void revin_kernel(const float* __restrict__ x, const float* __restrict__ rw,
                  const float* __restrict__ rb, float* __restrict__ xn,
                  float* __restrict__ meanb, float* __restrict__ stdb)
{
  __shared__ float red[256];
  const int bf = blockIdx.x;
  const int b = bf / F_, f = bf % F_;
  const int tid = threadIdx.x;
  float v = x[((long)b * L_ + tid) * F_ + f];
  red[tid] = v; __syncthreads();
  for (int s = 128; s > 0; s >>= 1) { if (tid < s) red[tid] += red[tid + s]; __syncthreads(); }
  float mean = red[0] * (1.0f / L_); __syncthreads();
  float d = v - mean;
  red[tid] = d * d; __syncthreads();
  for (int s = 128; s > 0; s >>= 1) { if (tid < s) red[tid] += red[tid + s]; __syncthreads(); }
  float stdev = sqrtf(red[0] * (1.0f / L_) + 1e-5f);
  xn[((long)b * L_ + tid) * F_ + f] = d / (stdev + 1e-5f) * rw[f] + rb[f];
  if (tid == 0) { meanb[bf] = mean; stdb[bf] = stdev; }
}

__global__ __launch_bounds__(256)
void embed_kernel(const int* __restrict__ tf,
                  const float* __restrict__ emo, const float* __restrict__ eda,
                  const float* __restrict__ ewd, const float* __restrict__ ehr,
                  const float* __restrict__ emi,
                  const float* __restrict__ Wc, const float* __restrict__ bc,
                  const float* __restrict__ Ws, const float* __restrict__ bs,
                  float* __restrict__ embs)
{
  int p = blockIdx.x * 256 + threadIdx.x;
  if (p >= ROWS_) return;
  const int* t = tf + (long)p * 5;
  float* o = embs + (long)p * 56;
  int mo = t[0] - 1, dy = t[1] - 1, wd = t[2], hr = t[3], mi = t[4];
#pragma unroll
  for (int j = 0; j < 8; ++j) {
    o[j]      = emo[mo * 8 + j];
    o[8 + j]  = eda[dy * 8 + j];
    o[16 + j] = ewd[wd * 8 + j];
    o[24 + j] = ehr[hr * 8 + j];
    o[32 + j] = emi[mi * 8 + j];
  }
  float hm  = (float)hr + (float)mi * (1.0f / 60.0f);
  float ang = 6.283185307179586f * hm * (1.0f / 24.0f);
  float tc = cosf(ang), ts = sinf(ang);
#pragma unroll
  for (int j = 0; j < 8; ++j) {
    o[40 + j] = tc * Wc[j] + bc[j];
    o[48 + j] = ts * Ws[j] + bs[j];
  }
}

__global__ __launch_bounds__(256)
void addpe_kernel(float* __restrict__ h, long n)
{
  for (long i = (long)blockIdx.x * 256 + threadIdx.x; i < n; i += (long)gridDim.x * 256) {
    int d = (int)(i % D_);
    int l = (int)((i / D_) % L_);
    int j = d >> 1;
    float div = expf((float)(2 * j) * (-9.210340371976184f / (float)D_));
    float ang = (float)l * div;
    h[i] += (d & 1) ? cosf(ang) : sinf(ang);
  }
}

__global__ __launch_bounds__(256)
void pad_kernel(const float* __restrict__ h, float* __restrict__ hp, long n)
{
  for (long i = (long)blockIdx.x * 256 + threadIdx.x; i < n; i += (long)gridDim.x * 256) {
    int d  = (int)(i % D_);
    long t = i / D_;
    int lp = (int)(t % LP_);
    int b  = (int)(t / LP_);
    int l  = lp - 3;
    hp[i] = (l >= 0 && l < L_) ? h[((long)b * L_ + l) * D_ + d] : 0.0f;
  }
}

__global__ __launch_bounds__(256)
void add_kernel(float* __restrict__ a, const float* __restrict__ b, long n)
{
  for (long i = (long)blockIdx.x * 256 + threadIdx.x; i < n; i += (long)gridDim.x * 256)
    a[i] += b[i];
}

__global__ __launch_bounds__(256)
void cvt_kernel(const float* __restrict__ s, bf16_t* __restrict__ d, long n)
{
  for (long i = (long)blockIdx.x * 256 + threadIdx.x; i < n; i += (long)gridDim.x * 256)
    d[i] = (bf16_t)s[i];
}

// src [NLc, Dout, Din, T] -> dst [NLc, T, Din, Dout] bf16
__global__ __launch_bounds__(256)
void cvt_convw_kernel(const float* __restrict__ s, bf16_t* __restrict__ d, int NLc, int T)
{
  long n = (long)NLc * T * D_ * D_;
  for (long i = (long)blockIdx.x * 256 + threadIdx.x; i < n; i += (long)gridDim.x * 256) {
    int  o  = (int)(i % D_);
    long t2 = i / D_;
    int  in = (int)(t2 % D_);
    long t3 = t2 / D_;
    int  t  = (int)(t3 % T);
    int  nl = (int)(t3 / T);
    d[i] = (bf16_t)s[(((long)nl * D_ + o) * D_ + in) * T + t];
  }
}

__global__ __launch_bounds__(256)
void final_kernel(const float* __restrict__ price, const float* __restrict__ vol,
                  const float* __restrict__ meanb, const float* __restrict__ stdb,
                  float* __restrict__ out)
{
  long n = (long)B_ * P_ * 48;
  for (long i = (long)blockIdx.x * 256 + threadIdx.x; i < n; i += (long)gridDim.x * 256) {
    int b = (int)(i / (P_ * 48));
    int rem = (int)(i % (P_ * 48));
    int p = rem / 48, j = rem % 48;
    float cs = stdb[b * F_ + 3], cm = meanb[b * F_ + 3];
    out[i] = (j < 24) ? price[((long)b * P_ + p) * 24 + j] * cs + cm
                      : vol[((long)b * P_ + p) * 24 + (j - 24)] * cs;
  }
}

// ---------------------------------------------------------------------------
// host side
// ---------------------------------------------------------------------------
static inline void run_gemm(hipStream_t st, const float* A, const bf16_t* Bm, float* C,
                            const float* bias, int M, int N, int K,
                            int lda, int ldb, int ldc, int flags,
                            bf16_t* Cbf = nullptr,
                            int aOuter = 0, int aLp = 0, int aOff = 0)
{
  dim3 grid((N + 127) / 128, (M + 127) / 128), blk(256);
  switch (flags) {
    case 1:  wmma_gemm_kernel<1><<<grid, blk, 0, st>>>(A, Bm, C, bias, Cbf, M, N, K, lda, ldb, ldc, aOuter, aLp, aOff); break;
    case 3:  wmma_gemm_kernel<3><<<grid, blk, 0, st>>>(A, Bm, C, bias, Cbf, M, N, K, lda, ldb, ldc, aOuter, aLp, aOff); break;
    case 5:  wmma_gemm_kernel<5><<<grid, blk, 0, st>>>(A, Bm, C, bias, Cbf, M, N, K, lda, ldb, ldc, aOuter, aLp, aOff); break;
    case 6:  wmma_gemm_kernel<6><<<grid, blk, 0, st>>>(A, Bm, C, bias, Cbf, M, N, K, lda, ldb, ldc, aOuter, aLp, aOff); break;
    case 10: wmma_gemm_kernel<10><<<grid, blk, 0, st>>>(A, Bm, C, bias, Cbf, M, N, K, lda, ldb, ldc, aOuter, aLp, aOff); break;
    default: wmma_gemm_kernel<2><<<grid, blk, 0, st>>>(A, Bm, C, bias, Cbf, M, N, K, lda, ldb, ldc, aOuter, aLp, aOff); break;
  }
}

extern "C" void kernel_launch(void* const* d_in, const int* in_sizes, int n_in,
                              void* d_out, int out_size, void* d_ws, size_t ws_size,
                              hipStream_t stream)
{
  (void)in_sizes; (void)n_in; (void)out_size; (void)ws_size;
  const float* x     = (const float*)d_in[0];
  const int*   tfeat = (const int*)d_in[1];
  const int*   sidx  = (const int*)d_in[2];
  const float* Wq    = (const float*)d_in[3];
  const float* bq    = (const float*)d_in[4];
  const float* Wk    = (const float*)d_in[5];
  const float* bk    = (const float*)d_in[6];
  const float* Wv    = (const float*)d_in[7];
  const float* bv    = (const float*)d_in[8];
  const float* Wo    = (const float*)d_in[9];
  const float* bo    = (const float*)d_in[10];
  const float* convW = (const float*)d_in[11];
  const float* convb = (const float*)d_in[12];
  const float* ln1w  = (const float*)d_in[13];
  const float* ln1b  = (const float*)d_in[14];
  const float* ln2w  = (const float*)d_in[15];
  const float* ln2b  = (const float*)d_in[16];
  const float* W1    = (const float*)d_in[17];
  const float* b1    = (const float*)d_in[18];
  const float* W2    = (const float*)d_in[19];
  const float* b2    = (const float*)d_in[20];
  const float* emo   = (const float*)d_in[21];
  const float* eda   = (const float*)d_in[22];
  const float* ewd   = (const float*)d_in[23];
  const float* ehr   = (const float*)d_in[24];
  const float* emi   = (const float*)d_in[25];
  const float* Wcos  = (const float*)d_in[26];
  const float* bcos  = (const float*)d_in[27];
  const float* Wsin  = (const float*)d_in[28];
  const float* bsin  = (const float*)d_in[29];
  const float* Wtime = (const float*)d_in[30];
  const float* btime = (const float*)d_in[31];
  const float* revw  = (const float*)d_in[32];
  const float* revb  = (const float*)d_in[33];
  const float* Wval  = (const float*)d_in[34];
  const float* bval  = (const float*)d_in[35];
  const float* decW3 = (const float*)d_in[36];
  const float* decb3 = (const float*)d_in[37];
  const float* decW5 = (const float*)d_in[38];
  const float* decb5 = (const float*)d_in[39];
  const float* decW7 = (const float*)d_in[40];
  const float* decb7 = (const float*)d_in[41];
  const float* Wproj = (const float*)d_in[42];
  const float* bproj = (const float*)d_in[43];
  const float* Wpr   = (const float*)d_in[44];
  const float* bpr   = (const float*)d_in[45];
  const float* Wvo   = (const float*)d_in[46];
  const float* bvo   = (const float*)d_in[47];
  float* out = (float*)d_out;

  // ---- workspace arena ----
  char* ws = (char*)d_ws;
  size_t off = 0;
  auto arena = [&](size_t bytes) -> void* {
    void* p = (void*)(ws + off);
    off = (off + bytes + 255) & ~(size_t)255;
    return p;
  };
  const size_t DD = (size_t)D_ * D_;
  bf16_t* wWq   = (bf16_t*)arena(NL_ * DD * 2);
  bf16_t* wWk   = (bf16_t*)arena(NL_ * DD * 2);
  bf16_t* wWv   = (bf16_t*)arena(NL_ * DD * 2);
  bf16_t* wWo   = (bf16_t*)arena(NL_ * DD * 2);
  bf16_t* wConv = (bf16_t*)arena((size_t)NL_ * 3 * DD * 2);
  bf16_t* wW1   = (bf16_t*)arena((size_t)NL_ * D_ * DFF_ * 2);
  bf16_t* wW2   = (bf16_t*)arena((size_t)NL_ * DFF_ * D_ * 2);
  bf16_t* wTime = (bf16_t*)arena((size_t)56 * D_ * 2);
  bf16_t* wVal  = (bf16_t*)arena((size_t)F_ * D_ * 2);
  bf16_t* wDec  = (bf16_t*)arena((size_t)15 * DD * 2);      // [3|5|7][in][out]
  bf16_t* wProj = (bf16_t*)arena((size_t)3 * D_ * D_ * 2);
  bf16_t* wPr   = (bf16_t*)arena((size_t)D_ * P_ * 2);
  bf16_t* wVo   = (bf16_t*)arena((size_t)D_ * P_ * 2);

  float*  hbuf  = (float*)arena((size_t)ROWS_ * D_ * 4);
  float*  Qb    = (float*)arena((size_t)ROWS_ * D_ * 4);
  float*  Kb    = (float*)arena((size_t)ROWS_ * D_ * 4);
  float*  Vb    = (float*)arena((size_t)ROWS_ * D_ * 4);
  bf16_t* Qbf   = (bf16_t*)arena((size_t)ROWS_ * D_ * 2);
  bf16_t* Kbf   = (bf16_t*)arena((size_t)ROWS_ * D_ * 2);
  float*  Cbuf  = (float*)arena((size_t)ROWS_ * D_ * 4);
  float*  Hpad  = (float*)arena((size_t)B_ * LP_ * D_ * 4);
  float*  Gbuf  = (float*)arena((size_t)ROWS_ * DFF_ * 4);  // also Dec3 (3D cols)
  float*  embs  = (float*)arena((size_t)ROWS_ * 56 * 4);
  float*  xn    = (float*)arena((size_t)ROWS_ * F_ * 4);
  float*  Mval  = (float*)arena((size_t)B_ * H_ * L_ * 4);
  float*  ctxB  = (float*)arena((size_t)B_ * D_ * 4);
  float*  Obuf  = (float*)arena((size_t)B_ * D_ * 4);
  float*  DecP  = (float*)arena((size_t)B_ * P_ * D_ * 4);
  float*  price = (float*)arena((size_t)B_ * P_ * P_ * 4);
  float*  vol   = (float*)arena((size_t)B_ * P_ * P_ * 4);
  float*  meanb = (float*)arena((size_t)B_ * F_ * 4);
  float*  stdb  = (float*)arena((size_t)B_ * F_ * 4);
  float*  Dec3  = Gbuf;                                     // reuse (decoder after FFN)

  auto gblk = [](long n) { return dim3((unsigned)((n + 255) / 256)); };

  // ---- weight conversion (fp32 -> bf16, conv transposed to [t][in][out]) ----
  cvt_kernel<<<gblk(NL_ * DD), 256, 0, stream>>>(Wq, wWq, NL_ * DD);
  cvt_kernel<<<gblk(NL_ * DD), 256, 0, stream>>>(Wk, wWk, NL_ * DD);
  cvt_kernel<<<gblk(NL_ * DD), 256, 0, stream>>>(Wv, wWv, NL_ * DD);
  cvt_kernel<<<gblk(NL_ * DD), 256, 0, stream>>>(Wo, wWo, NL_ * DD);
  cvt_convw_kernel<<<gblk((long)NL_ * 3 * DD), 256, 0, stream>>>(convW, wConv, NL_, 3);
  cvt_kernel<<<gblk((long)NL_ * D_ * DFF_), 256, 0, stream>>>(W1, wW1, (long)NL_ * D_ * DFF_);
  cvt_kernel<<<gblk((long)NL_ * DFF_ * D_), 256, 0, stream>>>(W2, wW2, (long)NL_ * DFF_ * D_);
  cvt_kernel<<<gblk(56L * D_), 256, 0, stream>>>(Wtime, wTime, 56L * D_);
  cvt_kernel<<<gblk((long)F_ * D_), 256, 0, stream>>>(Wval, wVal, (long)F_ * D_);
  cvt_convw_kernel<<<gblk(3L * DD), 256, 0, stream>>>(decW3, wDec, 1, 3);
  cvt_convw_kernel<<<gblk(5L * DD), 256, 0, stream>>>(decW5, wDec + 3 * DD, 1, 5);
  cvt_convw_kernel<<<gblk(7L * DD), 256, 0, stream>>>(decW7, wDec + 8 * DD, 1, 7);
  cvt_kernel<<<gblk(3L * D_ * D_), 256, 0, stream>>>(Wproj, wProj, 3L * D_ * D_);
  cvt_kernel<<<gblk((long)D_ * P_), 256, 0, stream>>>(Wpr, wPr, (long)D_ * P_);
  cvt_kernel<<<gblk((long)D_ * P_), 256, 0, stream>>>(Wvo, wVo, (long)D_ * P_);

  // ---- RevIN + embeddings + input projection ----
  revin_kernel<<<B_ * F_, 256, 0, stream>>>(x, revw, revb, xn, meanb, stdb);
  embed_kernel<<<gblk(ROWS_), 256, 0, stream>>>(tfeat, emo, eda, ewd, ehr, emi,
                                                Wcos, bcos, Wsin, bsin, embs);
  run_gemm(stream, embs, wTime, hbuf, btime, ROWS_, D_, 56, 56, D_, D_, GF_BIAS);
  run_gemm(stream, xn, wVal, hbuf, bval, ROWS_, D_, F_, F_, D_, D_, GF_ACCUM | GF_BIAS);
  addpe_kernel<<<gblk((long)ROWS_ * D_), 256, 0, stream>>>(hbuf, (long)ROWS_ * D_);

  // ---- encoder layers ----
  for (int i = 0; i < NL_; ++i) {
    run_gemm(stream, hbuf, wWq + i * DD, Qb, bq + i * D_, ROWS_, D_, D_, D_, D_, D_,
             GF_BIAS | GF_BF16, Qbf);
    run_gemm(stream, hbuf, wWk + i * DD, Kb, bk + i * D_, ROWS_, D_, D_, D_, D_, D_,
             GF_BIAS | GF_BF16, Kbf);
    run_gemm(stream, hbuf, wWv + i * DD, Vb, bv + i * D_, ROWS_, D_, D_, D_, D_, D_, GF_BIAS);

    score_kernel<<<B_ * H_, 256, 0, stream>>>(Qbf, Kbf, sidx + (long)i * S_, Mval);
    attend_kernel<<<B_ * H_, 256, 0, stream>>>(Mval, Qb, Kb, Vb, ctxB);

    run_gemm(stream, ctxB, wWo + i * DD, Obuf, bo + i * D_, B_, D_, D_, D_, D_, D_, GF_BIAS);
    ln_kernel<<<ROWS_, 256, 0, stream>>>(hbuf, Obuf, ln1w + i * D_, ln1b + i * D_, 2);

    pad_kernel<<<gblk((long)B_ * LP_ * D_), 256, 0, stream>>>(hbuf, Hpad, (long)B_ * LP_ * D_);
    for (int t = 0; t < 3; ++t)
      run_gemm(stream, Hpad, wConv + ((size_t)(i * 3 + t)) * DD, Cbuf, convb + i * D_,
               ROWS_, D_, D_, D_, D_, D_, (t == 0) ? GF_BIAS : GF_ACCUM,
               nullptr, L_, LP_, 2 + t);
    add_kernel<<<gblk((long)ROWS_ * D_), 256, 0, stream>>>(hbuf, Cbuf, (long)ROWS_ * D_);

    run_gemm(stream, hbuf, wW1 + (size_t)i * D_ * DFF_, Gbuf, b1 + i * DFF_,
             ROWS_, DFF_, D_, D_, DFF_, DFF_, GF_BIAS | GF_GELU);
    run_gemm(stream, Gbuf, wW2 + (size_t)i * DFF_ * D_, Cbuf, b2 + i * D_,
             ROWS_, D_, DFF_, DFF_, D_, D_, GF_BIAS);
    ln_kernel<<<ROWS_, 256, 0, stream>>>(hbuf, Cbuf, ln2w + i * D_, ln2b + i * D_, 1);
  }

  // ---- decoder: three gelu(conv) into Dec3 column slices, then proj + heads ----
  pad_kernel<<<gblk((long)B_ * LP_ * D_), 256, 0, stream>>>(hbuf, Hpad, (long)B_ * LP_ * D_);
  const int     Ts[3]    = {3, 5, 7};
  const size_t  wo3[3]   = {0, 3 * DD, 8 * DD};
  const float*  dbias[3] = {decb3, decb5, decb7};
  for (int c = 0; c < 3; ++c) {
    int T = Ts[c], pad = T / 2;
    for (int t = 0; t < T; ++t) {
      int flags = ((t == 0) ? GF_BIAS : GF_ACCUM) | ((t == T - 1) ? GF_GELU : 0);
      run_gemm(stream, Hpad, wDec + wo3[c] + (size_t)t * DD, Dec3 + c * D_, dbias[c],
               ROWS_, D_, D_, D_, D_, 3 * D_, flags, nullptr, L_, LP_, 3 - pad + t);
    }
  }
  // slice last P rows per batch and project 3D -> D
  run_gemm(stream, Dec3, wProj, DecP, bproj, B_ * P_, D_, 3 * D_,
           3 * D_, D_, D_, GF_BIAS, nullptr, P_, L_, L_ - P_);
  run_gemm(stream, DecP, wPr, price, bpr, B_ * P_, P_, D_, D_, P_, P_, GF_BIAS);
  run_gemm(stream, DecP, wVo, vol,   bvo, B_ * P_, P_, D_, D_, P_, P_, GF_BIAS);

  final_kernel<<<gblk((long)B_ * P_ * 48), 256, 0, stream>>>(price, vol, meanb, stdb, out);
}
